// FavorPlusAttention_49916109914491
// MI455X (gfx1250) — compile-verified, matrix operand order checked by reading, not run on hardware
//
#include <hip/hip_runtime.h>
#include <hip/hip_bf16.h>
#include <math.h>

// ---------------------------------------------------------------------------
// FAVOR+ causal linear attention for gfx1250 (MI455X), bf16 WMMA everywhere.
//   B=4, L=4096, D_MODEL=256, M_FEAT=512.
//   - phi(x) via tiled WMMA GEMM + exp epilogue (bf16 features in d_ws)
//   - causal scan as chunked linear attention: all matmuls on
//     v_wmma_f32_16x16x32_bf16, KV state f32-resident in LDS,
//     next-chunk operands streamed with GLOBAL_LOAD_ASYNC_TO_LDS_B128
//     (ASYNCcnt) into double buffers.
// ---------------------------------------------------------------------------

typedef __bf16 v16bf __attribute__((ext_vector_type(16)));
typedef __bf16 v8bf  __attribute__((ext_vector_type(8)));
typedef float  v8f   __attribute__((ext_vector_type(8)));

#define D_MODEL 256
#define M_FEAT  512
#define SEQ_L   4096
#define NBATCH  4
#define CHUNK   32          // rows per causal chunk
#define DV      32          // D-slice per workgroup in the scan kernel
#define NCHUNK  (SEQ_L / CHUNK)
#define EPS_F   1e-6f

// CDNA5 async global->LDS path (ASYNCcnt). Fall back to sync copies if the
// builtin is not exposed by this toolchain (also covers the host-side pass).
#if defined(__AMDGCN__) && __has_builtin(__builtin_amdgcn_global_load_async_to_lds_b128)
#define USE_ASYNC_LDS 1
#else
#define USE_ASYNC_LDS 0
#endif

__device__ __forceinline__ void copy16_g2lds(const void* g, void* l) {
#if USE_ASYNC_LDS
    // Builtin signature (from the compiler diagnostic): the pointee type is a
    // 16-byte int vector; global source is addrspace(1), LDS dest addrspace(3).
    typedef int b128 __attribute__((vector_size(16)));
    typedef __attribute__((address_space(1))) b128* gp_t;
    typedef __attribute__((address_space(3))) b128* lp_t;
    __builtin_amdgcn_global_load_async_to_lds_b128(
        (gp_t)g, (lp_t)l, /*offset=*/0, /*cpol=*/0);
#else
    *(uint4*)l = *(const uint4*)g;
#endif
}

__device__ __forceinline__ void wait_async_lds() {
#if USE_ASYNC_LDS
    asm volatile("s_wait_asynccnt 0x0" ::: "memory");
#endif
}

__device__ __forceinline__ v8f zero_v8f() {
    v8f z = {0.f, 0.f, 0.f, 0.f, 0.f, 0.f, 0.f, 0.f};
    return z;
}

// Load a 16x32 bf16 WMMA fragment (A-layout, or B-layout from a transposed
// matrix) from row-major storage. Per the CDNA5 ISA 16-bit A layout:
//   element e of lane: row = lane%16, k = e%8 + (e/8)*16 + (lane/16)*8
// so each lane needs two contiguous 8-element (16 B) loads.
__device__ __forceinline__ v16bf load_frag(const __bf16* base, int ld, int lane) {
    const int r  = lane & 15;
    const int hi = lane >> 4;
    const __bf16* p = base + r * ld + hi * 8;
    v8bf lo = *(const v8bf*)(p);
    v8bf hh = *(const v8bf*)(p + 16);
    v16bf f;
#pragma unroll
    for (int i = 0; i < 8; ++i) { f[i] = lo[i]; f[i + 8] = hh[i]; }
    return f;
}

__device__ __forceinline__ v8f wmma_bf16(v16bf a, v16bf b, v8f c) {
    return __builtin_amdgcn_wmma_f32_16x16x32_bf16(
        /*neg_a=*/false, a, /*neg_b=*/false, b,
        /*c_mod=*/(short)0, c, /*reuse_a=*/false, /*reuse_b=*/false);
}

// ---------------------------------------------------------------------------
// Kernel 1: phi(x) = exp(xW/sqrt(d) - 0.5*||x||^2) / sqrt(M), output bf16.
// Workgroup: 64 rows x 128 cols, 4 waves, K=256. Optionally also writes the
// transposed copy (per batch, [M][L]) needed by the scan's state update.
// ---------------------------------------------------------------------------
__global__ __launch_bounds__(128, 1)
void favor_phi_kernel(const float* __restrict__ X, const float* __restrict__ W,
                      __bf16* __restrict__ out_rm, __bf16* __restrict__ out_tr,
                      int write_tr)
{
    __shared__ __align__(16) __bf16 sX[64 * D_MODEL];     // 32 KB
    __shared__ __align__(16) __bf16 sWT[128 * D_MODEL];   // 64 KB (W^T slice)
    __shared__ float sSq[64];

    const int tid  = threadIdx.x;
    const int lane = tid & 31;
    const int wid  = tid >> 5;
    const int r0   = blockIdx.x * 64;     // row tile
    const int n0g  = blockIdx.y * 128;    // col tile

    // stage X (f32 -> bf16)
    {
        const float4* xg = (const float4*)(X + (size_t)r0 * D_MODEL);
#pragma unroll
        for (int i = 0; i < 32; ++i) {
            int idx4 = tid + i * 128;               // 4096 float4 total
            int e = idx4 * 4;
            float4 v = xg[idx4];
            sX[e + 0] = (__bf16)v.x; sX[e + 1] = (__bf16)v.y;
            sX[e + 2] = (__bf16)v.z; sX[e + 3] = (__bf16)v.w;
        }
    }
    // per-row 0.5*||x||^2 (from f32 source for accuracy)
    if (tid < 64) {
        const float* xr = X + (size_t)(r0 + tid) * D_MODEL;
        float s = 0.f;
        for (int k = 0; k < D_MODEL; k += 4) {
            float4 v = *(const float4*)(xr + k);
            s += v.x * v.x + v.y * v.y + v.z * v.z + v.w * v.w;
        }
        sSq[tid] = 0.5f * s;
    }
    // stage W^T slice: sWT[n][k] = W[k][n0g+n]
    for (int i = 0; i < 256; ++i) {
        int idx = tid + i * 128;                    // 32768 elems
        int n = idx >> 8;
        int k = idx & 255;
        sWT[idx] = (__bf16)W[(size_t)k * M_FEAT + n0g + n];
    }
    __syncthreads();

    const int m0 = wid * 16;                        // wave's 16-row slice
    v8f acc[8];
#pragma unroll
    for (int t = 0; t < 8; ++t) acc[t] = zero_v8f();

    for (int kk = 0; kk < D_MODEL; kk += 32) {
        v16bf a = load_frag(&sX[m0 * D_MODEL + kk], D_MODEL, lane);
#pragma unroll
        for (int t = 0; t < 8; ++t) {
            v16bf b = load_frag(&sWT[(t * 16) * D_MODEL + kk], D_MODEL, lane);
            acc[t] = wmma_bf16(a, b, acc[t]);
        }
    }

    const float invSqrtD = 0.0625f;                  // 1/sqrt(256)
    const float invSqrtM = 0.044194173824159216f;    // 1/sqrt(512)
    const int nIdx = lane & 15;
    const int mb   = (lane >> 4) * 8;
#pragma unroll
    for (int t = 0; t < 8; ++t) {
#pragma unroll
        for (int r = 0; r < 8; ++r) {
            int row = m0 + mb + r;
            int col = t * 16 + nIdx;
            float v = __expf(acc[t][r] * invSqrtD - sSq[row]) * invSqrtM;
            __bf16 bv = (__bf16)v;
            int gr = r0 + row, gc = n0g + col;
            out_rm[(size_t)gr * M_FEAT + gc] = bv;
            if (write_tr) {
                int b = gr >> 12;                    // row / 4096
                int l = gr & 4095;
                out_tr[((size_t)b * M_FEAT + gc) * SEQ_L + l] = bv;
            }
        }
    }
}

// ---------------------------------------------------------------------------
// Kernel 2: chunked causal linear-attention scan.
// grid = (D_MODEL/DV, NBATCH); one workgroup owns (batch, 32-wide D slice),
// keeps KV state (512x32 f32 + bf16 transpose) resident in LDS, and walks the
// 128 chunks of 32 rows. All matmuls are v_wmma_f32_16x16x32_bf16; next-chunk
// Q/K/K^T tiles are streamed into double buffers with async global->LDS.
// ---------------------------------------------------------------------------
__global__ __launch_bounds__(128, 1)
void favor_scan_kernel(const __bf16* __restrict__ qp, const __bf16* __restrict__ kp,
                       const __bf16* __restrict__ kpT, const float* __restrict__ V,
                       float* __restrict__ out)
{
    __shared__ __align__(16) __bf16 sQ [2][CHUNK * M_FEAT];  // 64 KB  Qc row-major
    __shared__ __align__(16) __bf16 sK [2][CHUNK * M_FEAT];  // 64 KB  Kc row-major (B^T for S)
    __shared__ __align__(16) __bf16 sKT[2][M_FEAT * CHUNK];  // 64 KB  Kc^T (A for KV update)
    __shared__ __align__(16) __bf16 sVT[DV * CHUNK];         //  2 KB  Vc^T (B^T for update/intra)
    __shared__ __align__(16) __bf16 sSm[CHUNK * CHUNK];      //  2 KB  masked S, bf16
    __shared__ __align__(16) float  sKV [M_FEAT * DV];       // 64 KB  KV state f32
    __shared__ __align__(16) __bf16 sKVT[DV * M_FEAT];       // 32 KB  KV^T bf16 (B^T for inter)
    __shared__ float sKst[M_FEAT];                           //  2 KB  k-state
    __shared__ float sDen[CHUNK];

    const int tid  = threadIdx.x;
    const int lane = tid & 31;
    const int wid  = tid >> 5;
    const int d0   = blockIdx.x * DV;
    const int bb   = blockIdx.y;

    // zero state
    for (int i = tid; i < M_FEAT * DV; i += 128) sKV[i] = 0.f;
    for (int i = tid; i < DV * M_FEAT; i += 128) sKVT[i] = (__bf16)0.f;
    for (int i = tid; i < M_FEAT; i += 128) sKst[i] = 0.f;
    __syncthreads();

    // issue one chunk's Q/K/K^T tiles (48 x b128 per thread) into buffer `buf`
    auto issue_chunk = [&](int c, int buf) {
        const size_t rb = (size_t)(bb * SEQ_L + c * CHUNK);
        const uint4* q4 = (const uint4*)(qp + rb * M_FEAT);
        const uint4* k4 = (const uint4*)(kp + rb * M_FEAT);
        uint4* dQ = (uint4*)sQ[buf];
        uint4* dK = (uint4*)sK[buf];
#pragma unroll
        for (int i = 0; i < 16; ++i) {              // 2048 x 16B each
            int s = tid + i * 128;
            copy16_g2lds(q4 + s, dQ + s);
            copy16_g2lds(k4 + s, dK + s);
        }
#pragma unroll
        for (int i = 0; i < 16; ++i) {              // kpT slice: 512 rows x 64 B
            int s = tid + i * 128;
            int m = s >> 2, part = s & 3;
            const uint4* src =
                (const uint4*)(kpT + ((size_t)(bb * M_FEAT + m)) * SEQ_L + c * CHUNK) + part;
            copy16_g2lds(src, (uint4*)(sKT[buf] + m * CHUNK) + part);
        }
    };

    const int nIdx = lane & 15;
    const int mb   = (lane >> 4) * 8;
    const int ti   = (wid >> 1) * 16;    // wave's 16x16 tile in the 32x32 outputs
    const int tj   = (wid & 1) * 16;

    issue_chunk(0, 0);                   // prologue: fill buffer 0

    for (int c = 0; c < NCHUNK; ++c) {
        const int buf = c & 1;
        const size_t rowbase = (size_t)(bb * SEQ_L + c * CHUNK);
        const __bf16* cQ  = sQ[buf];
        const __bf16* cK  = sK[buf];
        const __bf16* cKT = sKT[buf];

        // ---- Vc^T staging (f32 -> bf16 transpose; tiny, synchronous) ----
#pragma unroll
        for (int i = 0; i < 8; ++i) {
            int s = tid + i * 128;                   // 1024 elems
            int j = s >> 5, d = s & 31;
            float v = V[(rowbase + j) * D_MODEL + d0 + d];
            sVT[d * CHUNK + j] = (__bf16)v;
        }

        // current buffer was issued a full chunk ago: drain ASYNCcnt, then
        // immediately issue the next chunk into the other buffer so its loads
        // overlap this chunk's WMMA work.
        wait_async_lds();
        if (c + 1 < NCHUNK) {
            issue_chunk(c + 1, buf ^ 1);
            __builtin_prefetch(qp + (rowbase + 2 * CHUNK) * M_FEAT + tid * 256, 0, 0);
            __builtin_prefetch(kp + (rowbase + 2 * CHUNK) * M_FEAT + tid * 256, 0, 0);
        }
        __syncthreads();

        // ---- S = Qc * Kc^T (K = 512), causal mask, store bf16 ----
        {
            v8f s = zero_v8f();
#pragma unroll
            for (int kk = 0; kk < M_FEAT; kk += 32) {
                v16bf a = load_frag(&cQ[ti * M_FEAT + kk], M_FEAT, lane);
                v16bf b = load_frag(&cK[tj * M_FEAT + kk], M_FEAT, lane);
                s = wmma_bf16(a, b, s);
            }
#pragma unroll
            for (int r = 0; r < 8; ++r) {
                int i = ti + mb + r, j = tj + nIdx;
                float v = (j <= i) ? s[r] : 0.f;
                sSm[i * CHUNK + j] = (__bf16)v;
            }
        }
        __syncthreads();

        // ---- denominator: rowsum(masked S) + eps ----
        if (tid < CHUNK) {
            float rs = 0.f;
            for (int j = 0; j < CHUNK; ++j) rs += (float)sSm[tid * CHUNK + j];
            sDen[tid] = rs + EPS_F;
        }
        __syncthreads();

        // ---- denominator: + q . k_state (4-way split, ds_add_f32) ----
        {
            int i = tid & 31, part = tid >> 5;
            float p = 0.f;
            for (int m = part * 128; m < part * 128 + 128; ++m)
                p += (float)cQ[i * M_FEAT + m] * sKst[m];
            atomicAdd(&sDen[i], p);
        }

        // ---- numerator: inter = Qc * KV (K=512), intra = mask(S) * Vc ----
        v8f o = zero_v8f();
#pragma unroll
        for (int kk = 0; kk < M_FEAT; kk += 32) {
            v16bf a = load_frag(&cQ[ti * M_FEAT + kk], M_FEAT, lane);
            v16bf b = load_frag(&sKVT[tj * M_FEAT + kk], M_FEAT, lane);
            o = wmma_bf16(a, b, o);
        }
        {
            v16bf a = load_frag(&sSm[ti * CHUNK], CHUNK, lane);
            v16bf b = load_frag(&sVT[tj * CHUNK], CHUNK, lane);
            o = wmma_bf16(a, b, o);
        }
        __syncthreads();   // sDen complete; sKVT reads done before state update

        // ---- epilogue: out = num / den ----
#pragma unroll
        for (int r = 0; r < 8; ++r) {
            int i = ti + mb + r, d = tj + nIdx;
            size_t l = rowbase + i;
            out[l * D_MODEL + d0 + d] = o[r] / sDen[i];
        }

        // ---- k_state += colsum(Kc) ----
        for (int m = tid; m < M_FEAT; m += 128) {
            float s = 0.f;
            for (int j = 0; j < CHUNK; ++j) s += (float)cK[j * M_FEAT + m];
            sKst[m] += s;
        }

        // ---- KV += Kc^T * Vc ; state loaded as the WMMA C operand ----
#pragma unroll
        for (int tr = 0; tr < 8; ++tr) {
            int m0u = wid * 128 + tr * 16;
#pragma unroll
            for (int tc = 0; tc < 2; ++tc) {
                int n0u = tc * 16;
                v8f accu;
#pragma unroll
                for (int r = 0; r < 8; ++r)
                    accu[r] = sKV[(m0u + mb + r) * DV + n0u + nIdx];
                v16bf a   = load_frag(&cKT[m0u * CHUNK], CHUNK, lane);
                v16bf bfr = load_frag(&sVT[n0u * CHUNK], CHUNK, lane);
                accu = wmma_bf16(a, bfr, accu);
#pragma unroll
                for (int r = 0; r < 8; ++r) {
                    int mm = m0u + mb + r, nn = n0u + nIdx;
                    sKV[mm * DV + nn]       = accu[r];
                    sKVT[nn * M_FEAT + mm]  = (__bf16)accu[r];
                }
            }
        }
        __syncthreads();
    }
}

// ---------------------------------------------------------------------------
// Launch
// ---------------------------------------------------------------------------
extern "C" void kernel_launch(void* const* d_in, const int* in_sizes, int n_in,
                              void* d_out, int out_size, void* d_ws, size_t ws_size,
                              hipStream_t stream) {
    (void)in_sizes; (void)n_in; (void)out_size; (void)ws_size;
    const float* Q = (const float*)d_in[0];   // (B,L,D)
    const float* K = (const float*)d_in[1];   // (B,L,D)
    const float* V = (const float*)d_in[2];   // (B,L,D)
    const float* W = (const float*)d_in[3];   // (D,M)
    float* out = (float*)d_out;               // (B,L,D) f32

    const size_t rows = (size_t)NBATCH * SEQ_L;        // 16384
    __bf16* qp  = (__bf16*)d_ws;                       // rows x M  (16 MB)
    __bf16* kp  = qp + rows * M_FEAT;                  // rows x M  (16 MB)
    __bf16* kpT = kp + rows * M_FEAT;                  // B x M x L (16 MB)

    dim3 gphi(rows / 64, M_FEAT / 128);                // (256, 4)
    favor_phi_kernel<<<gphi, 128, 0, stream>>>(Q, W, qp, nullptr, 0);
    favor_phi_kernel<<<gphi, 128, 0, stream>>>(K, W, kp, kpT, 1);

    dim3 gscan(D_MODEL / DV, NBATCH);                  // (8, 4)
    favor_scan_kernel<<<gscan, 128, 0, stream>>>(qp, kp, kpT, V, out);
}
